// VoxelPruningSparseEncoder_22342419874282
// MI455X (gfx1250) — compile-verified, matrix operand order checked by reading, not run on hardware
//
#include <hip/hip_runtime.h>
#include <hip/hip_bf16.h>
#include <hip/hip_fp16.h>

// ---------------------------------------------------------------------------
// VoxelPruningSparseEncoder for MI455X (gfx1250, wave32, WMMA)
//
// Gather-bound workload: features kept f16 (halves L2 gather bytes, buffer is
// L2-resident), all GEMMs via v_wmma_f32_16x16x32_f16. Conv tap loop is
// software-pipelined: A-gather and B-staging for tap t+1 are issued before the
// WMMAs of tap t (double-buffered LDS for B, register ping-pong for A), plus
// global_prefetch for tap t+2 weights. Hash-table neighbors, histogram top-k.
// ---------------------------------------------------------------------------

typedef __attribute__((ext_vector_type(16))) _Float16 v16h;
typedef __attribute__((ext_vector_type(8)))  float    v8f;
typedef __attribute__((ext_vector_type(4)))  unsigned u32x4;

union Frag { u32x4 q[2]; v16h h; };   // 32 bytes = one lane's A or B fragment

#define IN_CH   5
#define SD      16
#define SH      256
#define SW      256
#define HBITS   18
#define HSIZE   (1 << HBITS)
#define HEMPTY  0xFFFFFFFFu
#define NBINS   4096

// ------------------------------ utility kernels ----------------------------

__global__ void zero_f32(float* p, int cnt) {
  int i = blockIdx.x * 256 + threadIdx.x;
  if (i < cnt) p[i] = 0.f;
}
__global__ void zero_i32(int* p, int cnt) {
  int i = blockIdx.x * 256 + threadIdx.x;
  if (i < cnt) p[i] = 0;
}

// ------------------------------ neighbor search ----------------------------

__device__ __forceinline__ unsigned hash_of(unsigned k) {
  return (k * 2654435761u) >> (32 - HBITS);
}

__global__ void hash_init(unsigned* hkey, int* hidx) {
  int i = blockIdx.x * 256 + threadIdx.x;
  if (i < HSIZE) { hkey[i] = HEMPTY; hidx[i] = 0x7FFFFFFF; }
}

__global__ void hash_insert(const int* __restrict__ coors,
                            unsigned* __restrict__ hkey, int* __restrict__ hidx,
                            int n) {
  int i = blockIdx.x * 256 + threadIdx.x;
  if (i >= n) return;
  int b = coors[i * 4 + 0], z = coors[i * 4 + 1];
  int y = coors[i * 4 + 2], x = coors[i * 4 + 3];
  unsigned key = (unsigned)(((b * SD + z) * SH + y) * SW + x);
  unsigned h = hash_of(key);
  for (;;) {
    unsigned prev = atomicCAS(&hkey[h], HEMPTY, key);
    if (prev == HEMPTY || prev == key) {
      atomicMin(&hidx[h], i);   // ties -> smallest original index (argsort stable)
      break;
    }
    h = (h + 1) & (HSIZE - 1);
  }
}

__global__ void probe_neighbors(const int* __restrict__ coors,
                                const unsigned* __restrict__ hkey,
                                const int* __restrict__ hidx,
                                int* __restrict__ nbr, int n) {
  int idx = blockIdx.x * 256 + threadIdx.x;
  if (idx >= n * 27) return;
  int i = idx / 27, t = idx - i * 27;
  int dz = t / 9 - 1, dy = (t / 3) % 3 - 1, dx = t % 3 - 1;
  int b = coors[i * 4 + 0];
  int z = coors[i * 4 + 1] + dz;
  int y = coors[i * 4 + 2] + dy;
  int x = coors[i * 4 + 3] + dx;
  int found = -1;
  if (z >= 0 && z < SD && y >= 0 && y < SH && x >= 0 && x < SW) {
    unsigned key = (unsigned)(((b * SD + z) * SH + y) * SW + x);
    unsigned h = hash_of(key);
    for (;;) {
      unsigned k = hkey[h];
      if (k == HEMPTY) break;
      if (k == key) { found = hidx[h]; break; }
      h = (h + 1) & (HSIZE - 1);
    }
  }
  nbr[idx] = found;
}

// ------------------------------ feature prep -------------------------------

// voxel_features f32 (N,5) -> f16 rows padded to 32 channels
__global__ void vf_prep(const float* __restrict__ vf, _Float16* __restrict__ f, int n) {
  int idx = blockIdx.x * 256 + threadIdx.x;
  if (idx >= n * 32) return;
  int i = idx >> 5, c = idx & 31;
  f[idx] = (_Float16)((c < IN_CH) ? vf[i * IN_CH + c] : 0.f);
}

// pack W (taps,CINR,COUTR) f32 -> f16 B fragments: [t][kt][nt][lane][16]
// fragment element e of lane: K = (e<8 ? e+hi*8 : 16+(e-8)+hi*8), N = lane&15
template<int CINR, int COUTR, int KT, int NT>
__global__ void pack_w(const float* __restrict__ W, _Float16* __restrict__ wp, int taps) {
  int total = taps * KT * NT * 512;
  int idx = blockIdx.x * 256 + threadIdx.x;
  if (idx >= total) return;
  int e    = idx & 15;
  int lane = (idx >> 4) & 31;
  int rest = idx >> 9;                 // (t*KT+kt)*NT + nt
  int nt   = rest % NT;
  int kt   = (rest / NT) % KT;
  int t    = rest / (NT * KT);
  int hi = lane >> 4, nn = lane & 15;
  int K = (e < 8) ? (e + hi * 8) : (16 + (e - 8) + hi * 8);
  int k = kt * 32 + K, c = nt * 16 + nn;
  float v = 0.f;
  if (k < CINR && c < COUTR) v = W[((size_t)t * CINR + k) * COUTR + c];
  wp[idx] = (_Float16)v;
}

// ------------------------- WMMA gather-conv kernel -------------------------
// Block = 256 threads = 8 waves; each wave owns a 16-voxel x COUT tile.
// !DENSE: 27 taps, software-pipelined (A regs ping-pong, B LDS double-buffer).
// SPSS  : prune gating + center-tap fallback. DENSE: plain row GEMM.
template<int KT, int NT, bool SPSS, bool DENSE>
__global__ __launch_bounds__(256)
void conv_wmma(const _Float16* __restrict__ feat,      // [n][KT*32] f16
               const _Float16* __restrict__ wpack,     // packed B fragments
               const int* __restrict__ nbr,            // [n][27] (-1 invalid)
               const unsigned char* __restrict__ maskb,// [n] prune mask (SPSS)
               float* __restrict__ out, int n)         // [n][NT*16] f32
{
  constexpr int CIN   = KT * 32;
  constexpr int COUT  = NT * 16;
  constexpr int FRAGS = KT * NT * 32;           // Frags per tap
  constexpr int LDSH  = DENSE ? 1 : 2 * FRAGS;  // double-buffered B stage
  __shared__ Frag lB[LDSH];

  const int lane = threadIdx.x & 31;
  const int wave = threadIdx.x >> 5;
  const int hi   = lane >> 4;
  const int li   = lane & 15;
  const int vox0 = (blockIdx.x * 8 + wave) * 16;
  const int arow = vox0 + li;          // A-fragment row this lane serves
  const u32x4 z4 = {0u, 0u, 0u, 0u};

  v8f acc[NT] = {};
  v8f ctr[SPSS ? NT : 1] = {};

  if constexpr (!DENSE) {
    constexpr int NQ   = KT * NT * 64;           // u32x4 per tap
    constexpr int NCPY = (NQ + 255) / 256;       // u32x4 per thread
    u32x4 breg[NCPY];
    Frag Acur[KT], Anxt[KT];
    const int* nrow = nbr + (size_t)arow * 27;

    auto stage_load = [&](int t) {               // global weights -> regs
      const u32x4* src = (const u32x4*)(wpack + (size_t)t * (KT * NT * 512));
#pragma unroll
      for (int i = 0; i < NCPY; ++i) {
        int k = (int)threadIdx.x + i * 256;
        if (k < NQ) breg[i] = src[k];
      }
    };
    auto stage_store = [&](int buf) {            // regs -> LDS buffer
      u32x4* dst = (u32x4*)(lB + buf * FRAGS);
#pragma unroll
      for (int i = 0; i < NCPY; ++i) {
        int k = (int)threadIdx.x + i * 256;
        if (k < NQ) dst[i * 256 + (int)threadIdx.x] = breg[i];
      }
    };
    auto gatherA = [&](int t, Frag* Af) {        // neighbor gather -> A frags
      int j = (arow < n) ? nrow[t] : -1;
      bool ok = j >= 0;
      if constexpr (SPSS) { if (ok) ok = maskb[j] != 0; }
#pragma unroll
      for (int kt = 0; kt < KT; ++kt) { Af[kt].q[0] = z4; Af[kt].q[1] = z4; }
      if (ok) {
        const u32x4* p = (const u32x4*)(feat + (size_t)j * CIN);
#pragma unroll
        for (int kt = 0; kt < KT; ++kt) {
          Af[kt].q[0] = p[kt * 4 + hi];       // K lo-chunk (0..7 / 8..15)
          Af[kt].q[1] = p[kt * 4 + 2 + hi];   // K hi-chunk (16..23 / 24..31)
        }
      }
    };

    // prologue: tap 0 staged + gathered
    stage_load(0);
    gatherA(0, Acur);
    stage_store(0);
    __syncthreads();

    for (int t = 0; t < 27; ++t) {
      const int cur = t & 1;
      const int nxt = cur ^ 1;
      if (t + 2 < 27) {
        // pull tap t+2 weight cachelines toward the WGP (global_prefetch_b8)
        if ((int)threadIdx.x * 128 < KT * NT * 512)
          __builtin_prefetch(wpack + (size_t)(t + 2) * (KT * NT * 512) +
                             (size_t)threadIdx.x * 128, 0, 3);
      }
      if (t + 1 < 27) {
        stage_load(t + 1);        // overlap: weight loads for t+1
        gatherA(t + 1, Anxt);     // overlap: random gather latency for t+1
      }
      const Frag* bf = lB + cur * FRAGS;
#pragma unroll
      for (int nt = 0; nt < NT; ++nt)
#pragma unroll
        for (int kt = 0; kt < KT; ++kt)
          acc[nt] = __builtin_amdgcn_wmma_f32_16x16x32_f16(
              false, Acur[kt].h, false, bf[(kt * NT + nt) * 32 + lane].h,
              (short)0, acc[nt], false, false);
      if (t + 1 < 27) {
        stage_store(nxt);
        __syncthreads();
#pragma unroll
        for (int kt = 0; kt < KT; ++kt) Acur[kt] = Anxt[kt];  // ping-pong
      }
    }
  }

  // Self pass: DENSE GEMM, or SPSS center fallback against W[CENTER_TAP=13]
  if constexpr (SPSS || DENSE) {
    Frag A[KT];
#pragma unroll
    for (int kt = 0; kt < KT; ++kt) { A[kt].q[0] = z4; A[kt].q[1] = z4; }
    if (arow < n) {
      const u32x4* p = (const u32x4*)(feat + (size_t)arow * CIN);
#pragma unroll
      for (int kt = 0; kt < KT; ++kt) {
        A[kt].q[0] = p[kt * 4 + hi];
        A[kt].q[1] = p[kt * 4 + 2 + hi];
      }
    }
    const Frag* wf = (const Frag*)(wpack + (size_t)(DENSE ? 0 : 13) * (KT * NT * 512));
#pragma unroll
    for (int nt = 0; nt < NT; ++nt)
#pragma unroll
      for (int kt = 0; kt < KT; ++kt) {
        v16h B = wf[(kt * NT + nt) * 32 + lane].h;
        if constexpr (DENSE)
          acc[nt] = __builtin_amdgcn_wmma_f32_16x16x32_f16(
              false, A[kt].h, false, B, (short)0, acc[nt], false, false);
        else
          ctr[nt] = __builtin_amdgcn_wmma_f32_16x16x32_f16(
              false, A[kt].h, false, B, (short)0, ctr[nt], false, false);
      }
  }

  // Store: C layout row M = r + 8*hi, col N = lane&15. SPSS selects center
  // for pruned-out rows.
#pragma unroll
  for (int r = 0; r < 8; ++r) {
    int v = vox0 + hi * 8 + r;
    if (v < n) {
      bool mv = true;
      if constexpr (SPSS) mv = maskb[v] != 0;
#pragma unroll
      for (int nt = 0; nt < NT; ++nt) {
        float val = acc[nt][r];
        if constexpr (SPSS) { if (!mv) val = ctr[nt][r]; }
        out[(size_t)v * COUT + nt * 16 + li] = val;
      }
    }
  }
}

// ------------------------------ BN / pruning -------------------------------

template<int C>
__global__ __launch_bounds__(256)
void bn_stats(const float* __restrict__ x, float* __restrict__ sums,
              float* __restrict__ sumsq, long total) {
  __shared__ float ls[C], lq[C];
  for (int c = threadIdx.x; c < C; c += 256) { ls[c] = 0.f; lq[c] = 0.f; }
  __syncthreads();
  for (long i = blockIdx.x * 256L + threadIdx.x; i < total; i += (long)gridDim.x * 256L) {
    float v = x[i];
    int c = (int)(i & (C - 1));
    atomicAdd(&ls[c], v);
    atomicAdd(&lq[c], v * v);
  }
  __syncthreads();
  for (int c = threadIdx.x; c < C; c += 256) {
    atomicAdd(&sums[c], ls[c]);
    atomicAdd(&sumsq[c], lq[c]);
  }
}

template<int C, int CPAD, bool RELU, bool HASRES, bool TO16, bool TO32>
__global__ __launch_bounds__(256)
void bn_apply(const float* __restrict__ conv, const float* __restrict__ sums,
              const float* __restrict__ sumsq, const float* __restrict__ g,
              const float* __restrict__ bb, const _Float16* __restrict__ res,
              _Float16* __restrict__ f16, float* __restrict__ f32, int n) {
  long total = (long)n * C;
  long idx = blockIdx.x * 256L + threadIdx.x;
  if (idx >= total) return;
  int c = (int)(idx & (C - 1));
  long v = idx / C;
  float nf = (float)n;
  float mean = sums[c] / nf;
  float var  = sumsq[c] / nf - mean * mean;
  float sc   = rsqrtf(var + 1e-3f);
  float val  = (conv[idx] - mean) * sc * g[c] + bb[c];
  if constexpr (HASRES) val += (float)res[v * CPAD + c];
  if constexpr (RELU) val = fmaxf(val, 0.f);
  if constexpr (TO16) {
    f16[v * CPAD + c] = (_Float16)val;
    if constexpr (CPAD > C) f16[v * CPAD + c + C] = (_Float16)0.f;  // zero pad lane
  }
  if constexpr (TO32) f32[idx] = val;
}

template<int C, int CPAD>
__global__ void imp_logit(const _Float16* __restrict__ f, const float* __restrict__ w,
                          float* __restrict__ lg, int n) {
  int i = blockIdx.x * 256 + threadIdx.x;
  if (i >= n) return;
  float s = 0.f;
#pragma unroll
  for (int c = 0; c < C; ++c) s += (float)f[(size_t)i * CPAD + c] * w[c];
  lg[i] = s;
}

__global__ void hist_fill(const float* __restrict__ lg, int* __restrict__ hist, int n) {
  int i = blockIdx.x * 256 + threadIdx.x;
  if (i >= n) return;
  int b = (int)((lg[i] + 16.f) * ((float)NBINS / 32.f));
  b = min(max(b, 0), NBINS - 1);
  atomicAdd(&hist[b], 1);
}

// kth-largest threshold (logit space; sigmoid monotonic) via top-down scan
__global__ void hist_select(const int* __restrict__ hist, float* __restrict__ th, int k) {
  if (threadIdx.x == 0 && blockIdx.x == 0) {
    int cum = 0;
    float t = -1e30f;
    for (int b = NBINS - 1; b >= 0; --b) {
      cum += hist[b];
      if (cum >= k) { t = -16.f + (float)b * (32.f / (float)NBINS); break; }
    }
    th[0] = t;
  }
}

template<int CPAD>
__global__ void fg_prep(const _Float16* __restrict__ f, const float* __restrict__ lg,
                        const float* __restrict__ th, _Float16* __restrict__ fg,
                        unsigned char* __restrict__ mk, int n) {
  int idx = blockIdx.x * 256 + threadIdx.x;
  if (idx >= n * CPAD) return;
  int i = idx / CPAD, c = idx - i * CPAD;
  float l = lg[i];
  float s = 1.f / (1.f + expf(-l));
  fg[idx] = (_Float16)((float)f[idx] * s);
  if (c == 0) mk[i] = (l >= th[0]) ? 1 : 0;
}

// ------------------------------- launcher ----------------------------------

extern "C" void kernel_launch(void* const* d_in, const int* in_sizes, int n_in,
                              void* d_out, int out_size, void* d_ws, size_t ws_size,
                              hipStream_t stream) {
  (void)n_in; (void)out_size; (void)ws_size;
  const int N = in_sizes[0] / IN_CH;

  const float* vf    = (const float*)d_in[0];
  const int*   coors = (const int*)d_in[1];
  // params in reference-dict insertion order:
  // in{W,g,b}, s*{W,imp,g,b} x7, out{W,g,b}, batch_size
  const float* inW = (const float*)d_in[2];
  const float* inG = (const float*)d_in[3];
  const float* inB = (const float*)d_in[4];
  struct SP { const float *W, *imp, *g, *b; };
  auto sp = [&](int base) {
    return SP{(const float*)d_in[base], (const float*)d_in[base + 1],
              (const float*)d_in[base + 2], (const float*)d_in[base + 3]};
  };
  SP s1 = sp(5), s2a = sp(9), s2b = sp(13), s2c = sp(17);
  SP s3a = sp(21), s4a = sp(25), s4b = sp(29);
  const float* outW = (const float*)d_in[33];
  const float* outG = (const float*)d_in[34];
  const float* outB = (const float*)d_in[35];

  // workspace arena
  char* base_p = (char*)d_ws;
  size_t off = 0;
  auto carve = [&](size_t bytes) -> char* {
    off = (off + 255) & ~(size_t)255;
    char* p = base_p + off;
    off += bytes;
    return p;
  };
  unsigned*      hkey  = (unsigned*)carve((size_t)HSIZE * 4);
  int*           hidx  = (int*)carve((size_t)HSIZE * 4);
  int*           nbr   = (int*)carve((size_t)N * 27 * 4);
  _Float16*      F0    = (_Float16*)carve((size_t)N * 64 * 2);
  _Float16*      F1    = (_Float16*)carve((size_t)N * 64 * 2);
  _Float16*      F2    = (_Float16*)carve((size_t)N * 64 * 2);
  _Float16*      FG    = (_Float16*)carve((size_t)N * 64 * 2);
  float*         cbuf  = (float*)carve((size_t)N * 128 * 4);
  float*         logit = (float*)carve((size_t)N * 4);
  unsigned char* mk    = (unsigned char*)carve((size_t)N);
  int*           hist  = (int*)carve((size_t)NBINS * 4);
  float*         th    = (float*)carve(256);
  float*         bs    = (float*)carve(128 * 4);
  float*         bq    = (float*)carve(128 * 4);
  _Float16*      wp    = (_Float16*)carve((size_t)27 * 2 * 4 * 1024);

  auto blk = [](long total) { return dim3((unsigned)((total + 255) / 256)); };
  const dim3 cgrid((unsigned)((N + 127) / 128));   // 128 voxels / block

  // ---- neighbor table ----
  hash_init<<<blk(HSIZE), 256, 0, stream>>>(hkey, hidx);
  hash_insert<<<blk(N), 256, 0, stream>>>(coors, hkey, hidx, N);
  probe_neighbors<<<blk((long)N * 27), 256, 0, stream>>>(coors, hkey, hidx, nbr, N);

  // ---- input subm conv: 5(->32) -> 16 ----
  vf_prep<<<blk((long)N * 32), 256, 0, stream>>>(vf, F2, N);
  pack_w<IN_CH, 16, 1, 1><<<blk(27L * 512), 256, 0, stream>>>(inW, wp, 27);
  conv_wmma<1, 1, false, false><<<cgrid, 256, 0, stream>>>(F2, wp, nbr, nullptr, cbuf, N);
  zero_f32<<<1, 256, 0, stream>>>(bs, 128);
  zero_f32<<<1, 256, 0, stream>>>(bq, 128);
  bn_stats<16><<<dim3(1024), 256, 0, stream>>>(cbuf, bs, bq, (long)N * 16);
  bn_apply<16, 32, true, false, true, false><<<blk((long)N * 16), 256, 0, stream>>>(
      cbuf, bs, bq, inG, inB, (const _Float16*)nullptr, F0, (float*)nullptr, N);

#define SPSS_LAYER(P, CINc, CPADIN, COUTc, CPADOUT, KTc, NTc, FIN, HASRES, RESPTR, FOUT)   \
  do {                                                                                     \
    imp_logit<CINc, CPADIN><<<blk(N), 256, 0, stream>>>(FIN, P.imp, logit, N);             \
    zero_i32<<<blk(NBINS), 256, 0, stream>>>(hist, NBINS);                                 \
    hist_fill<<<blk(N), 256, 0, stream>>>(logit, hist, N);                                 \
    hist_select<<<1, 32, 0, stream>>>(hist, th, N / 2);                                    \
    fg_prep<CPADIN><<<blk((long)N * CPADIN), 256, 0, stream>>>(FIN, logit, th, FG, mk, N); \
    pack_w<CINc, COUTc, KTc, NTc><<<blk(27L * KTc * NTc * 512), 256, 0, stream>>>(         \
        P.W, wp, 27);                                                                      \
    conv_wmma<KTc, NTc, true, false><<<cgrid, 256, 0, stream>>>(FG, wp, nbr, mk, cbuf, N); \
    zero_f32<<<1, 256, 0, stream>>>(bs, 128);                                              \
    zero_f32<<<1, 256, 0, stream>>>(bq, 128);                                              \
    bn_stats<COUTc><<<dim3(1024), 256, 0, stream>>>(cbuf, bs, bq, (long)N * COUTc);        \
    bn_apply<COUTc, CPADOUT, true, HASRES, true, false>                                    \
        <<<blk((long)N * COUTc), 256, 0, stream>>>(cbuf, bs, bq, P.g, P.b, RESPTR, FOUT,   \
                                                   (float*)nullptr, N);                    \
  } while (0)

  // ---- SPSS stack (feature ping-pong F0/F1/F2; residuals kept live) ----
  SPSS_LAYER(s1,  16, 32, 16, 32, 1, 1, F0, false, (const _Float16*)nullptr, F1);
  SPSS_LAYER(s2a, 16, 32, 32, 32, 1, 2, F1, false, (const _Float16*)nullptr, F0); // idn = F0
  SPSS_LAYER(s2b, 32, 32, 32, 32, 1, 2, F0, false, (const _Float16*)nullptr, F1);
  SPSS_LAYER(s2c, 32, 32, 32, 32, 1, 2, F1, true,  (const _Float16*)F0,      F2); // +idn
  SPSS_LAYER(s3a, 32, 32, 64, 64, 1, 4, F2, false, (const _Float16*)nullptr, F0); // idn = F0
  SPSS_LAYER(s4a, 64, 64, 64, 64, 2, 4, F0, false, (const _Float16*)nullptr, F1);
  SPSS_LAYER(s4b, 64, 64, 64, 64, 2, 4, F1, true,  (const _Float16*)F0,      F2); // +idn
#undef SPSS_LAYER

  // ---- final dense GEMM 64 -> 128, BN, ReLU -> d_out (f32) ----
  pack_w<64, 128, 2, 8><<<blk(1L * 2 * 8 * 512), 256, 0, stream>>>(outW, wp, 1);
  conv_wmma<2, 8, false, true><<<cgrid, 256, 0, stream>>>(F2, wp, nullptr, nullptr, cbuf, N);
  zero_f32<<<1, 256, 0, stream>>>(bs, 128);
  zero_f32<<<1, 256, 0, stream>>>(bq, 128);
  bn_stats<128><<<dim3(1024), 256, 0, stream>>>(cbuf, bs, bq, (long)N * 128);
  bn_apply<128, 128, true, false, false, true><<<blk((long)N * 128), 256, 0, stream>>>(
      cbuf, bs, bq, outG, outB, (const _Float16*)nullptr, (_Float16*)nullptr,
      (float*)d_out, N);
}
